// CostVolume_45964740002516
// MI455X (gfx1250) — matrix operand order repaired and checked by reading, compiled, not run on hardware
//
#include <hip/hip_runtime.h>

typedef __attribute__((ext_vector_type(16))) _Float16 v16h;
typedef __attribute__((ext_vector_type(8)))  float    v8f;

#define BATCH 8
#define HH    128
#define WW    192
#define CC    128
#define MDISP 4
#define NEG_SLOPE 0.1f
#define WTILES (WW / 16)          // 12
#define WAVES_PER_BLOCK 8

// A-matrix fragment (16x32 f16, M x K), per the CDNA5 ISA 16-bit A layout:
// lanes 0-15 (hi=0): halves 0..7 = K c0+0..7, halves 8..15 = K c0+16..23
// lanes 16-31 (hi=1): halves 0..7 = K c0+8..15, halves 8..15 = K c0+24..31
__device__ __forceinline__ v16h pack_a(const float* __restrict__ row, int c0, int hi) {
    const float* p = row + c0 + (hi ? 8 : 0);
    float4 x0 = *reinterpret_cast<const float4*>(p);
    float4 x1 = *reinterpret_cast<const float4*>(p + 4);
    float4 x2 = *reinterpret_cast<const float4*>(p + 16);
    float4 x3 = *reinterpret_cast<const float4*>(p + 20);
    v16h f;
    f[0]  = (_Float16)x0.x; f[1]  = (_Float16)x0.y; f[2]  = (_Float16)x0.z; f[3]  = (_Float16)x0.w;
    f[4]  = (_Float16)x1.x; f[5]  = (_Float16)x1.y; f[6]  = (_Float16)x1.z; f[7]  = (_Float16)x1.w;
    f[8]  = (_Float16)x2.x; f[9]  = (_Float16)x2.y; f[10] = (_Float16)x2.z; f[11] = (_Float16)x2.w;
    f[12] = (_Float16)x3.x; f[13] = (_Float16)x3.y; f[14] = (_Float16)x3.z; f[15] = (_Float16)x3.w;
    return f;
}

// B-matrix fragment (32x16 f16, K x N), per the CDNA5 ISA B layout pattern:
// lane group g = lane>>4 holds K = c0 + g*8 .. +7 and c0 + 16 + g*8 .. +7 (matches A),
// column N = lane & 15. We read row-major warped rows (N = warped pixel).
__device__ __forceinline__ v16h pack_b(const float* __restrict__ row, int c0, int g) {
    const float* p = row + c0 + g * 8;
    float4 x0 = *reinterpret_cast<const float4*>(p);
    float4 x1 = *reinterpret_cast<const float4*>(p + 4);
    float4 x2 = *reinterpret_cast<const float4*>(p + 16);
    float4 x3 = *reinterpret_cast<const float4*>(p + 20);
    v16h f;
    f[0]  = (_Float16)x0.x; f[1]  = (_Float16)x0.y; f[2]  = (_Float16)x0.z; f[3]  = (_Float16)x0.w;
    f[4]  = (_Float16)x1.x; f[5]  = (_Float16)x1.y; f[6]  = (_Float16)x1.z; f[7]  = (_Float16)x1.w;
    f[8]  = (_Float16)x2.x; f[9]  = (_Float16)x2.y; f[10] = (_Float16)x2.z; f[11] = (_Float16)x2.w;
    f[12] = (_Float16)x3.x; f[13] = (_Float16)x3.y; f[14] = (_Float16)x3.z; f[15] = (_Float16)x3.w;
    return f;
}

__global__ __launch_bounds__(256) void cost_volume_wmma(
        const float* __restrict__ x1,
        const float* __restrict__ warped,
        float* __restrict__ out) {
    // Per-wave LDS slice: 16 pixels x 32 band columns (cols w0-8 .. w0+23)
    __shared__ float lds[WAVES_PER_BLOCK][16][32];

    const int lane = threadIdx.x & 31;
    const int wave = threadIdx.x >> 5;
    const int tile = blockIdx.x * WAVES_PER_BLOCK + wave;

    const int wt = tile % WTILES;
    const int h  = (tile / WTILES) % HH;
    const int b  = tile / (WTILES * HH);
    const int w0 = wt * 16;

    const int ln = lane & 15;   // pixel-in-tile (A row) / B column
    const int hi = lane >> 4;   // lane group

    // ---- Load A fragments once (x1 tile 16x128), reused across all 9 sh ----
    const float* arow = x1 + (((size_t)b * HH + h) * WW + (w0 + ln)) * CC;
    v16h afr[4];
#pragma unroll
    for (int kk = 0; kk < 4; ++kk) afr[kk] = pack_a(arow, kk * 32, hi);

    // B column bases for the two band tiles
    const int col1 = w0 - 8 + ln;          // tile 1: cols w0-8 .. w0+7
    const int col2 = w0 + 8 + ln;          // tile 2: cols w0+8 .. w0+23
    const bool v1 = (col1 >= 0);           // col1 < WW always (max 183)
    const bool v2 = (col2 < WW);           // col2 >= 8 always
    const float scale = 1.0f / (float)CC;

    v16h zf = {};

    for (int sh = -MDISP; sh <= MDISP; ++sh) {
        const int hh = h + sh;
        const bool rowOK = (hh >= 0) && (hh < HH);   // wave-uniform

        if (rowOK) {
            const float* wbase = warped + (((size_t)b * HH + hh) * WW) * CC;
            const float* brow1 = wbase + (size_t)col1 * CC;
            const float* brow2 = wbase + (size_t)col2 * CC;

            v8f acc1 = {};
            v8f acc2 = {};
#pragma unroll
            for (int kk = 0; kk < 4; ++kk) {
                v16h bf1 = v1 ? pack_b(brow1, kk * 32, hi) : zf;
                acc1 = __builtin_amdgcn_wmma_f32_16x16x32_f16(
                        false, afr[kk], false, bf1, (short)0, acc1, false, false);
                v16h bf2 = v2 ? pack_b(brow2, kk * 32, hi) : zf;
                acc2 = __builtin_amdgcn_wmma_f32_16x16x32_f16(
                        false, afr[kk], false, bf2, (short)0, acc2, false, false);
            }

            // C/D layout: vgpr r, lane l -> row M = r + 8*(l>>4), col N = l&15
#pragma unroll
            for (int r = 0; r < 8; ++r) {
                lds[wave][r + 8 * hi][ln]      = acc1[r];
                lds[wave][r + 8 * hi][16 + ln] = acc2[r];
            }
        }
        __syncthreads();

        // ---- Band extraction + mean + LeakyReLU + store ----
        // lane group 0 handles sw indices 0..4, group 1 handles 5..8 (pixel m = ln)
        {
            const int m = ln;
            float* obase = out + (((size_t)b * HH + h) * WW + (w0 + m)) * 81
                               + (size_t)(sh + MDISP) * 9;
#pragma unroll
            for (int i = 0; i < 5; ++i) {
                int swi = hi * 5 + i;
                if (swi < 9) {
                    // band column j = (w0+m+sw) - (w0-8) = m + sw + 8 = m + swi + 4
                    float v = rowOK ? lds[wave][m][m + swi + 4] : 0.0f;
                    v *= scale;
                    v = (v >= 0.0f) ? v : NEG_SLOPE * v;
                    __builtin_nontemporal_store(v, obase + swi);
                }
            }
        }
        __syncthreads();
    }
}

extern "C" void kernel_launch(void* const* d_in, const int* in_sizes, int n_in,
                              void* d_out, int out_size, void* d_ws, size_t ws_size,
                              hipStream_t stream) {
    const float* x1     = (const float*)d_in[0];
    const float* warped = (const float*)d_in[1];
    float* out          = (float*)d_out;

    const int tiles  = BATCH * HH * WTILES;          // 12288
    const int blocks = tiles / WAVES_PER_BLOCK;      // 1536
    hipLaunchKernelGGL(cost_volume_wmma, dim3(blocks), dim3(256), 0, stream,
                       x1, warped, out);
}
